// EMAVectorQuantizer_28338194219131
// MI455X (gfx1250) — matrix operand order repaired and checked by reading, not compiled, not run on hardware
//
#include <hip/hip_runtime.h>
#include <hip/hip_bf16.h>
#include <math.h>

// ---------------------------------------------------------------------------
// VQ: N=16384 rows x D=256 vs K=8192 codes.
// argmin ||fz - e||^2 with both l2-normalized == argmax fz . e^T
// Dominant work: 16384x8192x256 GEMM -> v_wmma_f32_16x16x32_f16.
// Codebook tiles staged into double-buffered LDS via async global->LDS loads
// (ASYNCcnt) when the toolchain exposes the gfx1250 builtins.
// ---------------------------------------------------------------------------

#define DDIM     256
#define KCODES   8192
#define NROWS    16384
#define MTILE    128          // rows per block (8 waves x 16 rows)
#define CHUNK    16           // codes per LDS-staged chunk
#define LDS_ROW  272          // halves per code row in LDS (256 + 16 pad = 544B, 32B aligned)

typedef __attribute__((ext_vector_type(16))) _Float16 v16h;
typedef __attribute__((ext_vector_type(8)))  _Float16 v8h;
typedef __attribute__((ext_vector_type(8)))  float    v8f;

#if defined(__gfx1250__) && __has_builtin(__builtin_amdgcn_global_load_async_to_lds_b128)
#define VQ_ASYNC 1
#else
#define VQ_ASYNC 0
#endif

#if VQ_ASYNC
// Parameter type per clang-22 diagnostic: non-const pointer to
// 'int __attribute__((vector_size(16)))' in AS(1) (global) / AS(3) (LDS).
typedef __attribute__((__vector_size__(4 * sizeof(int)))) int vq_i4;
typedef __attribute__((address_space(1))) vq_i4* vq_gptr;   // global int4*
typedef __attribute__((address_space(3))) vq_i4* vq_lptr;   // LDS int4*
#define TO_GLB(p) ((vq_gptr)(p))
#define TO_LDS(p) ((vq_lptr)(p))
static __device__ __forceinline__ void vq_wait_async() {
#if __has_builtin(__builtin_amdgcn_s_wait_asynccnt)
    __builtin_amdgcn_s_wait_asynccnt(0);
#else
    asm volatile("s_wait_asynccnt 0x0" ::: "memory");
#endif
}
#endif

// ---------------------------------------------------------------------------
// Kernel 1: l2-normalize rows of z and embed_w, emit f16 copies.
// One wave (32 lanes) per row, 8 floats per lane.
// ---------------------------------------------------------------------------
__global__ __launch_bounds__(256) void vq_normalize_kernel(
    const float* __restrict__ z, const float* __restrict__ ew,
    _Float16* __restrict__ fz16, _Float16* __restrict__ e16)
{
    const int row  = blockIdx.x * 8 + (threadIdx.x >> 5);
    const int lane = threadIdx.x & 31;
    const float* src;
    _Float16*    dst;
    if (row < NROWS) {
        src = z    + (size_t)row * DDIM;
        dst = fz16 + (size_t)row * DDIM;
    } else {
        const int r = row - NROWS;
        if (r >= KCODES) return;
        src = ew  + (size_t)r * DDIM;
        dst = e16 + (size_t)r * DDIM;
    }
    const float4 x0 = *(const float4*)(src + lane * 8);
    const float4 x1 = *(const float4*)(src + lane * 8 + 4);
    float ss = x0.x*x0.x + x0.y*x0.y + x0.z*x0.z + x0.w*x0.w
             + x1.x*x1.x + x1.y*x1.y + x1.z*x1.z + x1.w*x1.w;
    #pragma unroll
    for (int m = 16; m >= 1; m >>= 1) ss += __shfl_xor(ss, m, 32);
    const float inv = 1.0f / fmaxf(sqrtf(ss), 1e-12f);
    v8h o;
    o[0] = (_Float16)(x0.x * inv); o[1] = (_Float16)(x0.y * inv);
    o[2] = (_Float16)(x0.z * inv); o[3] = (_Float16)(x0.w * inv);
    o[4] = (_Float16)(x1.x * inv); o[5] = (_Float16)(x1.y * inv);
    o[6] = (_Float16)(x1.z * inv); o[7] = (_Float16)(x1.w * inv);
    *(v8h*)(dst + lane * 8) = o;
}

// ---------------------------------------------------------------------------
// Kernel 2: WMMA GEMM + running argmax.
// Block: 256 threads = 8 waves; wave w owns rows [blk*128 + w*16, +16).
// A fragments (16x256 f16) resident in 8 v16h regs per lane.
// Codebook chunk (16 codes x 256 dims) staged in LDS, shared by all 8 waves.
// Async path: double-buffered LDS, one barrier per chunk.
// ---------------------------------------------------------------------------
__global__ __launch_bounds__(256) void vq_argmax_kernel(
    const _Float16* __restrict__ fz16, const _Float16* __restrict__ e16,
    int* __restrict__ idx_out)
{
#if VQ_ASYNC
    __shared__ __align__(32) _Float16 lb[2][CHUNK * LDS_ROW];
#else
    __shared__ __align__(32) _Float16 lb[1][CHUNK * LDS_ROW];
#endif

    const int tid   = threadIdx.x;
    const int lane  = tid & 31;
    const int wave  = tid >> 5;
    const int l16   = lane & 15;
    const int hsel  = lane >> 4;              // 0: lanes 0-15, 1: lanes 16-31
    const int rowBase = blockIdx.x * MTILE + wave * 16;
    const int m     = rowBase + l16;

    // --- load resident A fragments, contiguous k order (k = 16*hsel + elem), same as B:
    //     lanes 0-15: elems 0..15 = K 0..15
    //     lanes16-31: elems 0..15 = K 16..31
    v16h a[8];
    const _Float16* arow = fz16 + (size_t)m * DDIM;
    #pragma unroll
    for (int kk = 0; kk < 8; ++kk) {
        const v8h lo = *(const v8h*)(arow + kk * 32 + hsel * 16);
        const v8h hi = *(const v8h*)(arow + kk * 32 + hsel * 16 + 8);
        v16h t;
        #pragma unroll
        for (int i = 0; i < 8; ++i) { t[i] = lo[i]; t[8 + i] = hi[i]; }
        a[kk] = t;
    }

    float best[8];
    int   bidx[8];
    #pragma unroll
    for (int r = 0; r < 8; ++r) { best[r] = -3.0e38f; bidx[r] = 0; }

    const int scode = tid >> 4;   // staging: thread -> (code, 32B part)
    const int spart = tid & 15;

#if VQ_ASYNC
    // prologue: async-stage chunk 0 into buffer 0
    {
        const _Float16* g = e16 + (size_t)scode * DDIM + spart * 16;
        _Float16*       l = &lb[0][scode * LDS_ROW + spart * 16];
        __builtin_amdgcn_global_load_async_to_lds_b128(TO_GLB(g), TO_LDS(l), 0, 0);
        __builtin_amdgcn_global_load_async_to_lds_b128(TO_GLB(g), TO_LDS(l), 16, 0);
    }
#endif

    int buf = 0;
    for (int cb = 0; cb < KCODES; cb += CHUNK) {
#if VQ_ASYNC
        vq_wait_async();       // my wave's async LDS writes complete
        __syncthreads();       // everyone's writes complete -> tile `buf` ready
        if (cb + CHUNK < KCODES) {
            const _Float16* g = e16 + (size_t)(cb + CHUNK + scode) * DDIM + spart * 16;
            _Float16*       l = &lb[buf ^ 1][scode * LDS_ROW + spart * 16];
            __builtin_amdgcn_global_load_async_to_lds_b128(TO_GLB(g), TO_LDS(l), 0, 0);
            __builtin_amdgcn_global_load_async_to_lds_b128(TO_GLB(g), TO_LDS(l), 16, 0);
        }
#else
        // stage 16 codes x 256 halves into LDS (each thread copies 32B)
        {
            const uint4* src = (const uint4*)(e16 + (size_t)(cb + scode) * DDIM + spart * 16);
            uint4* dst = (uint4*)(&lb[0][scode * LDS_ROW + spart * 16]);
            dst[0] = src[0];
            dst[1] = src[1];
        }
        __syncthreads();
        if (cb + CHUNK < KCODES)  // gfx1250 global_prefetch_b8
            __builtin_prefetch(e16 + (size_t)(cb + CHUNK + scode) * DDIM + spart * 16, 0, 1);
#endif

        // B 32x16 f16 layout: lanes 0-15 hold K 0..15 (col N=lane),
        // lanes 16-31 hold K 16..31 (col N=lane-16)
        const _Float16* lbase = &lb[buf][l16 * LDS_ROW + hsel * 16];
        v8f c = {};
        #pragma unroll
        for (int kk = 0; kk < 8; ++kk) {
            const v16h b = *(const v16h*)(lbase + kk * 32);
            c = __builtin_amdgcn_wmma_f32_16x16x32_f16(
                    false, a[kk], false, b, (short)0, c, false, false);
        }

        // C layout: VGPR r -> row rowBase + hsel*8 + r, col N = l16
        const int code = cb + l16;
        #pragma unroll
        for (int r = 0; r < 8; ++r) {
            if (c[r] > best[r]) { best[r] = c[r]; bidx[r] = code; }
        }
#if VQ_ASYNC
        buf ^= 1;
#else
        __syncthreads();
#endif
    }

    // argmax across the 16 N-lanes of each half-wave (tie -> lowest index)
    #pragma unroll
    for (int r = 0; r < 8; ++r) {
        float v  = best[r];
        int   ix = bidx[r];
        #pragma unroll
        for (int mk = 8; mk >= 1; mk >>= 1) {
            const float ov = __shfl_xor(v, mk, 32);
            const int   oi = __shfl_xor(ix, mk, 32);
            if (ov > v || (ov == v && oi < ix)) { v = ov; ix = oi; }
        }
        bidx[r] = ix;
    }
    if (l16 == 0) {
        const int rb = rowBase + hsel * 8;
        #pragma unroll
        for (int r = 0; r < 8; ++r) idx_out[rb + r] = bidx[r];
    }
}

// ---------------------------------------------------------------------------
// Kernel 3: gather raw codebook rows, write quantized_st = z + (q - z),
// write idx as float, per-row commitment-loss partial sums.
// One wave per row.
// ---------------------------------------------------------------------------
__global__ __launch_bounds__(256) void vq_gather_loss_kernel(
    const float* __restrict__ z, const float* __restrict__ ew,
    const int* __restrict__ idx, float* __restrict__ out,
    float* __restrict__ partials)
{
    const int row  = blockIdx.x * 8 + (threadIdx.x >> 5);
    const int lane = threadIdx.x & 31;
    if (row >= NROWS) return;
    const int code = idx[row];

    const float* q  = ew + (size_t)code * DDIM;
    const float* zr = z  + (size_t)row  * DDIM;
    const float4 q0 = *(const float4*)(q  + lane * 8);
    const float4 q1 = *(const float4*)(q  + lane * 8 + 4);
    const float4 z0 = *(const float4*)(zr + lane * 8);
    const float4 z1 = *(const float4*)(zr + lane * 8 + 4);

    // straight-through output: z + (q - z) elementwise (matches reference fp math)
    float4 o0, o1;
    o0.x = z0.x + (q0.x - z0.x); o0.y = z0.y + (q0.y - z0.y);
    o0.z = z0.z + (q0.z - z0.z); o0.w = z0.w + (q0.w - z0.w);
    o1.x = z1.x + (q1.x - z1.x); o1.y = z1.y + (q1.y - z1.y);
    o1.z = z1.z + (q1.z - z1.z); o1.w = z1.w + (q1.w - z1.w);
    *(float4*)(out + (size_t)row * DDIM + lane * 8)     = o0;
    *(float4*)(out + (size_t)row * DDIM + lane * 8 + 4) = o1;

    float sq = q0.x*q0.x + q0.y*q0.y + q0.z*q0.z + q0.w*q0.w
             + q1.x*q1.x + q1.y*q1.y + q1.z*q1.z + q1.w*q1.w;
    float sz = z0.x*z0.x + z0.y*z0.y + z0.z*z0.z + z0.w*z0.w
             + z1.x*z1.x + z1.y*z1.y + z1.z*z1.z + z1.w*z1.w;
    #pragma unroll
    for (int mk = 16; mk >= 1; mk >>= 1) {
        sq += __shfl_xor(sq, mk, 32);
        sz += __shfl_xor(sz, mk, 32);
    }
    const float iq = 1.0f / fmaxf(sqrtf(sq), 1e-12f);
    const float iz = 1.0f / fmaxf(sqrtf(sz), 1e-12f);

    float ds = 0.0f;
    {
        float d;
        d = q0.x*iq - z0.x*iz; ds += d*d;
        d = q0.y*iq - z0.y*iz; ds += d*d;
        d = q0.z*iq - z0.z*iz; ds += d*d;
        d = q0.w*iq - z0.w*iz; ds += d*d;
        d = q1.x*iq - z1.x*iz; ds += d*d;
        d = q1.y*iq - z1.y*iz; ds += d*d;
        d = q1.z*iq - z1.z*iz; ds += d*d;
        d = q1.w*iq - z1.w*iz; ds += d*d;
    }
    #pragma unroll
    for (int mk = 16; mk >= 1; mk >>= 1) ds += __shfl_xor(ds, mk, 32);

    if (lane == 0) {
        partials[row] = ds;
        out[(size_t)NROWS * DDIM + row] = (float)code;  // idx output (float dtype)
    }
}

// ---------------------------------------------------------------------------
// Kernel 4: deterministic reduction of per-row partials -> vq_loss scalar.
// ---------------------------------------------------------------------------
__global__ __launch_bounds__(256) void vq_finalize_kernel(
    const float* __restrict__ partials, float* __restrict__ out)
{
    __shared__ float sb[256];
    float s = 0.0f;
    for (int i = threadIdx.x; i < NROWS; i += 256) s += partials[i];
    sb[threadIdx.x] = s;
    __syncthreads();
    #pragma unroll
    for (int st = 128; st > 0; st >>= 1) {
        if (threadIdx.x < st) sb[threadIdx.x] += sb[threadIdx.x + st];
        __syncthreads();
    }
    if (threadIdx.x == 0)
        out[(size_t)NROWS * DDIM + NROWS] = 0.25f * sb[0] / (float)((size_t)NROWS * DDIM);
}

// ---------------------------------------------------------------------------
// Launch
// ---------------------------------------------------------------------------
extern "C" void kernel_launch(void* const* d_in, const int* in_sizes, int n_in,
                              void* d_out, int out_size, void* d_ws, size_t ws_size,
                              hipStream_t stream)
{
    (void)in_sizes; (void)n_in; (void)out_size; (void)ws_size;

    const float* z  = (const float*)d_in[0];   // [16384, 256]
    const float* ew = (const float*)d_in[1];   // [8192, 256]
    float* out = (float*)d_out;

    char* ws = (char*)d_ws;
    _Float16* fz16     = (_Float16*)ws;                                // 8 MB
    _Float16* e16      = (_Float16*)(ws + (size_t)NROWS * DDIM * 2);   // 4 MB
    int*      idx      = (int*)     (ws + (size_t)NROWS * DDIM * 2
                                        + (size_t)KCODES * DDIM * 2);  // 64 KB
    float*    partials = (float*)   (ws + (size_t)NROWS * DDIM * 2
                                        + (size_t)KCODES * DDIM * 2
                                        + (size_t)NROWS * 4);          // 64 KB

    vq_normalize_kernel<<<(NROWS + KCODES) / 8, 256, 0, stream>>>(z, ew, fz16, e16);
    vq_argmax_kernel<<<NROWS / MTILE, 256, 0, stream>>>(fz16, e16, idx);
    vq_gather_loss_kernel<<<NROWS / 8, 256, 0, stream>>>(z, ew, idx, out, partials);
    vq_finalize_kernel<<<1, 256, 0, stream>>>(partials, out);
}
